// Self_Attention_1d_18949395710129
// MI455X (gfx1250) — matrix operand order repaired and checked by reading, not compiled, not run on hardware
//
#include <hip/hip_runtime.h>
#include <hip/hip_bf16.h>

typedef __bf16 bf16;
typedef __attribute__((ext_vector_type(16))) __bf16 v16bf;
typedef __attribute__((ext_vector_type(8)))  float  v8f;
typedef unsigned int v4u __attribute__((ext_vector_type(4)));
typedef int v4i __attribute__((ext_vector_type(4)));
typedef int v8i __attribute__((ext_vector_type(8)));

constexpr int kS = 4096;
constexpr int kB = 2;
constexpr int kD = 512;

#if defined(__has_include)
#if __has_include(<hip/amd_detail/amd_gfx1250_TDM.h>)
#define TDM_6ARG 1
#else
#define TDM_6ARG 0
#endif
#else
#define TDM_6ARG 0
#endif

union FragU { uint4 u[2]; v16bf v; };

// ---- WMMA wrapper: D = A(16x32 bf16) * B(32x16 bf16) + C(16x16 f32) ----
__device__ inline v8f wmma_bf16(v16bf a, v16bf b, v8f c) {
  return __builtin_amdgcn_wmma_f32_16x16x32_bf16(
      /*neg_a=*/false, a, /*neg_b=*/false, b,
      /*c_mod=*/(short)0, c, /*reuse_a=*/false, /*reuse_b=*/false);
}

// A-fragment (16x32, MxK) from row-major bf16 tile, per ISA 7.12.2 layout.
__device__ inline v16bf loadA_bf16(const bf16* tile, int ld, int lane) {
  const int m = lane & 15, h = lane >> 4;
  const bf16* p = tile + m * ld + h * 8;
  FragU f;
  f.u[0] = *(const uint4*)(p);        // K = h*8 .. h*8+7
  f.u[1] = *(const uint4*)(p + 16);   // K = 16 + h*8 ..
  return f.v;
}

// B-fragment (32x16, KxN) sourced from a ROW-MAJOR [N][K] matrix (B^T).
__device__ inline v16bf loadB_bf16(const bf16* tileT, int ld, int lane) {
  const int n = lane & 15, g = lane >> 4;
  const bf16* p = tileT + n * ld + g * 8;
  FragU f;
  f.u[0] = *(const uint4*)(p);        // K = 8g + 0..7
  f.u[1] = *(const uint4*)(p + 16);   // K = 16 + 8g + 0..7
  return f.v;
}

// fp32-source fragments (projection inputs), converted on the fly.
__device__ inline v16bf loadA_f32(const float* tile, int ld, int lane) {
  const int m = lane & 15, h = lane >> 4;
  const float* p = tile + m * ld + h * 8;
  v16bf f;
#pragma unroll
  for (int j = 0; j < 8; ++j) f[j] = (bf16)p[j];
#pragma unroll
  for (int j = 0; j < 8; ++j) f[8 + j] = (bf16)p[16 + j];
  return f;
}
__device__ inline v16bf loadB_f32(const float* tileT, int ld, int lane) {
  const int n = lane & 15, g = lane >> 4;
  const float* p = tileT + n * ld + g * 8;
  v16bf f;
#pragma unroll
  for (int j = 0; j < 16; ++j) f[j] = (bf16)p[j + (j & 8)];
  return f;
}

// ---- Tensor Data Mover: 2D tile load Global -> LDS (bf16 elements) ----
// Descriptor per CDNA5 ISA ch.8: group0 = {count, lds_addr, global_addr, type=2},
// group1 = {data_size=2B, tensor_dim0/1, tile_dim0/1, tensor_dim0_stride}.
__device__ inline void tdm_load_2d_bf16(unsigned lds_byte_off, const void* gaddr,
                                        unsigned tensor_d0, unsigned tensor_d1,
                                        unsigned stride0_elems,
                                        unsigned tile_d0, unsigned tile_d1) {
  unsigned long long ga = (unsigned long long)gaddr;
  v4u g0;
  g0[0] = 1u;                                            // count=1, user mode
  g0[1] = lds_byte_off;                                  // lds_addr (bytes)
  g0[2] = (unsigned)(ga & 0xFFFFFFFFu);                  // global_addr[31:0]
  g0[3] = (unsigned)((ga >> 32) & 0x01FFFFFFu) | (2u << 30);  // addr[56:32] | type=2
  v8i g1;
  g1[0] = (int)(1u << 16);                               // data_size=1 -> 2 bytes
  g1[1] = (int)((tensor_d0 & 0xFFFFu) << 16);            // tensor_dim0 [79:48]
  g1[2] = (int)((tensor_d0 >> 16) | ((tensor_d1 & 0xFFFFu) << 16));  // dim0 hi | dim1 lo
  g1[3] = (int)((tensor_d1 >> 16) | (tile_d0 << 16));    // dim1 hi | tile_dim0 [127:112]
  g1[4] = (int)(tile_d1 & 0xFFFFu);                      // tile_dim1 [143:128], tile_dim2=0
  g1[5] = (int)stride0_elems;                            // tensor_dim0_stride [191:160]
  g1[6] = 0;                                             // stride0 hi / stride1 lo
  g1[7] = 0;
  v4i z4 = {0, 0, 0, 0};
#if TDM_6ARG
  v8i z8 = {0, 0, 0, 0, 0, 0, 0, 0};
  __builtin_amdgcn_tensor_load_to_lds(g0, g1, z4, z4, z8, 0);
#else
  __builtin_amdgcn_tensor_load_to_lds(g0, g1, z4, z4, 0);
#endif
}

// ============================================================================
// Projection: Out = X @ W^T + b  (rows r = s*B + b match X memory layout)
// transposeOut==0: Out[b][s][d] (bf16);  ==1: Out[b][d][s] (bf16, for V^T)
// ============================================================================
__global__ __launch_bounds__(256) void proj_kernel(
    const float* __restrict__ X,     // [S*B][D]
    const float* __restrict__ W,     // [D][D] row-major [out][in]
    const float* __restrict__ bias,  // [D]
    bf16* __restrict__ Out, int transposeOut) {
  const int lane = threadIdx.x & 31;
  const int wid  = (blockIdx.x * blockDim.x + threadIdx.x) >> 5;
  const int colTiles = kD / 64;              // 8
  const int r0 = (wid / colTiles) * 16;
  const int c0 = (wid % colTiles) * 64;

  v8f acc[4] = {};
#pragma unroll 4
  for (int k0 = 0; k0 < kD; k0 += 32) {
    v16bf a = loadA_f32(X + r0 * kD + k0, kD, lane);
#pragma unroll
    for (int nt = 0; nt < 4; ++nt) {
      v16bf b = loadB_f32(W + (c0 + nt * 16) * kD + k0, kD, lane);
      acc[nt] = wmma_bf16(a, b, acc[nt]);
    }
  }

  const int n  = lane & 15;
  const int mb = (lane >> 4) * 8;  // C layout: VGPR r -> m = r + 8*(lane/16)
#pragma unroll
  for (int nt = 0; nt < 4; ++nt) {
    const int col = c0 + nt * 16 + n;
    const float bv = bias[col];
#pragma unroll
    for (int r = 0; r < 8; ++r) {
      const int row = r0 + mb + r;           // row = s*B + b
      const int s = row / kB, bb = row % kB;
      const float val = acc[nt][r] + bv;
      const long idx = transposeOut ? ((long)(bb * kD + col) * kS + s)
                                    : ((long)(bb * kS + s) * kD + col);
      Out[idx] = (bf16)val;
    }
  }
}

// ============================================================================
// Fused tanh-attention with TDM double-buffered K / V^T staging.
// LDS: Q 32KB | K[2] 2x64KB | Vt[2] 2x64KB | Ps 4KB  = 292KB (<320KB/WGP)
// Block = 8 waves covering 32 s-rows x 512 d of one batch.
// ============================================================================
constexpr unsigned QS_OFF = 0;                        // 32*512*2 = 32768
constexpr unsigned KS_OFF = 32768;                    // 2 x 64*512*2 = 131072
constexpr unsigned KBUF   = 64 * 512 * 2;             // 65536
constexpr unsigned VS_OFF = KS_OFF + 2 * KBUF;        // 163840
constexpr unsigned VBUF   = 512 * 64 * 2;             // 65536
constexpr unsigned PS_OFF = VS_OFF + 2 * VBUF;        // 294912
constexpr unsigned SMEM_BYTES = PS_OFF + 2 * 16 * 64 * 2;  // 299008

__global__ __launch_bounds__(256) void attn_kernel(
    const bf16* __restrict__ Qb,     // [B][S][D]
    const bf16* __restrict__ Kb,     // [B][S][D]
    const bf16* __restrict__ Vt,     // [B][D][S]
    const float* __restrict__ value, // [S][B][D]
    const float* __restrict__ gamma_p,
    float* __restrict__ out)         // [S][B][D]
{
  extern __shared__ char smem[];
  bf16* Qs = (bf16*)(smem + QS_OFF);   // [32][512]
  bf16* Ps = (bf16*)(smem + PS_OFF);   // [2][16][64] tanh scores

  const int bat  = blockIdx.y;
  const int s0   = blockIdx.x * 32;
  const int tid  = threadIdx.x;
  const int lane = tid & 31;
  const int wid  = tid >> 5;
  const int ssub = wid >> 2;           // 0..1
  const int dquad = wid & 3;           // 0..3
  const int d0   = dquad * 128;
  const float gamma = *gamma_p;
  const int rounds = kS / 64;

  // Prologue: wave 0 kicks TDM for round 0 tiles into buffer 0.
  if (wid == 0) {
    const bf16* kg = Kb + ((long)bat * kS + 0) * kD;
    tdm_load_2d_bf16(KS_OFF, kg, kD, 64, kD, kD, 64);          // K: 512-wide x 64 rows
    const bf16* vg = Vt + ((long)bat * kD + 0) * kS + 0;
    tdm_load_2d_bf16(VS_OFF, vg, 64, kD, kS, 64, kD);          // V^T: 64-wide x 512 rows
  }

  // Stage Q tile (32x512) once, all threads.
  {
    const uint4* src4 = (const uint4*)(Qb + ((long)bat * kS + s0) * kD);
    uint4* dst4 = (uint4*)Qs;
    for (int i = tid; i < 32 * 512 / 8; i += 256) dst4[i] = src4[i];
  }

  v8f accO[8] = {};  // 16 s-rows x 128 d-cols, fp32

  for (int it = 0; it < rounds; ++it) {
    const int buf = it & 1;
    if (wid == 0) __builtin_amdgcn_s_wait_tensorcnt(0);  // round-it tiles landed
    __syncthreads();  // publish tiles; prior round's readers of other buf done

    // Issue next round's TDM into the other buffer (overlaps this round's math).
    if (wid == 0 && it + 1 < rounds) {
      const int tn = (it + 1) * 64;
      const int nbuf = (it + 1) & 1;
      const bf16* kg = Kb + ((long)bat * kS + tn) * kD;
      tdm_load_2d_bf16(KS_OFF + nbuf * KBUF, kg, kD, 64, kD, kD, 64);
      const bf16* vg = Vt + ((long)bat * kD + 0) * kS + tn;
      tdm_load_2d_bf16(VS_OFF + nbuf * VBUF, vg, 64, kD, kS, 64, kD);
    }

    const bf16* Ks  = (bf16*)(smem + KS_OFF + buf * KBUF);   // [64][512]
    const bf16* Vts = (bf16*)(smem + VS_OFF + buf * VBUF);   // [512][64]

    // ---- QK^T: this wave computes a distinct 16 s x 16 t score block ----
    v8f sc = {};
    const bf16* qtile = Qs + ssub * 16 * 512;
    const bf16* ktile = Ks + dquad * 16 * 512;   // B^T rows = K rows
#pragma unroll
    for (int k0 = 0; k0 < 512; k0 += 32) {
      v16bf a = loadA_bf16(qtile + k0, 512, lane);
      v16bf b = loadB_bf16(ktile + k0, 512, lane);
      sc = wmma_bf16(a, b, sc);
    }

    // tanh, reshape C-layout -> row-major bf16 scores in LDS (A-matrix for PV)
    {
      const int n  = lane & 15;
      const int mb = (lane >> 4) * 8;
      bf16* prow = Ps + ssub * 16 * 64;
#pragma unroll
      for (int r = 0; r < 8; ++r)
        prow[(mb + r) * 64 + dquad * 16 + n] = (bf16)tanhf(sc[r]);
    }
    __syncthreads();

    // ---- P(16x64) x V(64x128): accumulate output ----
    const bf16* ptile = Ps + ssub * 16 * 64;
#pragma unroll
    for (int kt = 0; kt < 2; ++kt) {
      v16bf a = loadA_bf16(ptile + kt * 32, 64, lane);
#pragma unroll
      for (int nt = 0; nt < 8; ++nt) {
        v16bf b = loadB_bf16(Vts + (d0 + nt * 16) * 64 + kt * 32, 64, lane);
        accO[nt] = wmma_bf16(a, b, accO[nt]);
      }
    }
  }

  // Epilogue: out[s][b][d] = gamma*acc + value[s][b][d]
  const int n  = lane & 15;
  const int mb = (lane >> 4) * 8;
#pragma unroll
  for (int nt = 0; nt < 8; ++nt) {
    const int d = d0 + nt * 16 + n;
#pragma unroll
    for (int r = 0; r < 8; ++r) {
      const int s = s0 + ssub * 16 + mb + r;
      const long idx = ((long)s * kB + bat) * kD + d;
      out[idx] = gamma * accO[nt][r] + value[idx];
    }
  }
}

extern "C" void kernel_launch(void* const* d_in, const int* in_sizes, int n_in,
                              void* d_out, int out_size, void* d_ws, size_t ws_size,
                              hipStream_t stream) {
  const float* query = (const float*)d_in[0];
  const float* key   = (const float*)d_in[1];
  const float* value = (const float*)d_in[2];
  const float* Wq    = (const float*)d_in[3];
  const float* bq    = (const float*)d_in[4];
  const float* Wk    = (const float*)d_in[5];
  const float* bk    = (const float*)d_in[6];
  const float* Wv    = (const float*)d_in[7];
  const float* bv    = (const float*)d_in[8];
  const float* gamma = (const float*)d_in[9];
  float* out = (float*)d_out;

  const size_t mat = (size_t)kB * kS * kD;  // 4 Mi elements (8 MB bf16)
  bf16* Qb  = (bf16*)d_ws;
  bf16* Kbf = Qb + mat;
  bf16* VtT = Kbf + mat;

  const int pblocks = (kS * kB / 16) * (kD / 64) / 8;  // 512
  proj_kernel<<<pblocks, 256, 0, stream>>>(query, Wq, bq, Qb, 0);
  proj_kernel<<<pblocks, 256, 0, stream>>>(key,   Wk, bk, Kbf, 0);
  proj_kernel<<<pblocks, 256, 0, stream>>>(value, Wv, bv, VtT, 1);

  attn_kernel<<<dim3(kS / 32, kB), 256, SMEM_BYTES, stream>>>(Qb, Kbf, VtT,
                                                              value, gamma, out);
}
